// raw_gru_s2s_60971355734180
// MI455X (gfx1250) — matrix operand order repaired and checked by reading, not compiled, not run on hardware
//
#include <hip/hip_runtime.h>
#include <hip/hip_bf16.h>

typedef __attribute__((ext_vector_type(16))) _Float16 v16h;
typedef __attribute__((ext_vector_type(8)))  _Float16 v8h;
typedef __attribute__((ext_vector_type(8)))  float    v8f;

#define GN      20000   // nodes
#define GE      320000  // edges
#define GG      128     // graphs
#define GHID    128
#define GHEADS  4
#define GIN     74
#define GCOLS   (GHEADS*GHID)   // 512
#define GEPS    1e-5f

// ---------- helpers: order-preserving float<->uint for atomic max ----------
__device__ __forceinline__ unsigned fenc(float f) {
    unsigned u = __float_as_uint(f);
    return (u & 0x80000000u) ? ~u : (u | 0x80000000u);
}
__device__ __forceinline__ float fdec(unsigned u) {
    return (u & 0x80000000u) ? __uint_as_float(u & 0x7FFFFFFFu)
                             : __uint_as_float(~u);
}

// ============================================================================
// Operand packing (one-shot per layer; all bounds logic lives here)
// ============================================================================
// X [M,K] f32 -> Xh [M,Kpad] f16, zero padded K..Kpad
__global__ void pack_x_f16(const float* __restrict__ X, _Float16* __restrict__ Xh,
                           int K, int Kpad) {
    int n = blockIdx.x;
    const float* xr = X + (size_t)n * K;
    _Float16* dr = Xh + (size_t)n * Kpad;
    for (int k = threadIdx.x; k < Kpad; k += blockDim.x)
        dr[k] = (k < K) ? (_Float16)xr[k] : (_Float16)0.0f;
}

// W [4,K,128] f32 -> Wp in B-fragment order: [ntile][kchunk][lane][16 halves]
// lane (r=lane&15, hh=lane>>4): column = ntile*16+r, half i -> k = kc+16*hh+i
__global__ void pack_w_f16(const float* __restrict__ W, _Float16* __restrict__ Wp,
                           int K, int Kpad) {
    int nch = Kpad >> 5;
    int total = 32 * nch * 512;           // ntile * chunk * (32 lanes * 16 halves)
    int t = blockIdx.x * blockDim.x + threadIdx.x;
    if (t >= total) return;
    int i    = t & 15;
    int lane = (t >> 4) & 31;
    int rest = t >> 9;
    int ch   = rest % nch;
    int nt   = rest / nch;
    int r = lane & 15, hh = lane >> 4;
    int gcol = nt * 16 + r;
    int head = gcol >> 7, c = gcol & 127;
    int k = (ch << 5) + 16 * hh + i;
    Wp[t] = (k < K) ? (_Float16)W[((size_t)head * K + k) * GHID + c]
                    : (_Float16)0.0f;
}

// ============================================================================
// WMMA GEMM: Z[M,512] = Xh[M,Kpad] @ W. One wave computes a 16x64 strip
// (4 accumulators sharing the A fragment). M = 1250 full 16-row tiles.
// Inner loop: 2x b128 A loads, 2x b128 B loads per tile, 4x v_wmma.
// ============================================================================
__global__ void gat_gemm_wmma(const _Float16* __restrict__ Xh,
                              const _Float16* __restrict__ Wp,
                              float* __restrict__ Z, int Mtiles, int Kpad) {
    const int wave = threadIdx.x >> 5;
    const int lane = threadIdx.x & 31;
    const int mtile = blockIdx.x * 8 + wave;
    if (mtile >= Mtiles) return;              // wave-uniform exit
    const int r  = lane & 15;
    const int hh = lane >> 4;
    const int nt0 = blockIdx.y * 4;           // 4 column tiles per wave
    const int nch = Kpad >> 5;

    const _Float16* xr = Xh + (size_t)(mtile * 16 + r) * Kpad;
    v8f acc[4] = {};

    for (int ch = 0; ch < nch; ++ch) {
        const int kc = ch << 5;
        v16h a = *(const v16h*)(xr + kc + 16 * hh);
#pragma unroll
        for (int t = 0; t < 4; ++t) {
            const _Float16* bp =
                Wp + (((size_t)(nt0 + t) * nch + ch) << 9) + lane * 16;
            v16h b = *(const v16h*)bp;
            acc[t] = __builtin_amdgcn_wmma_f32_16x16x32_f16(
                false, a, false, b, (short)0, acc[t], false, false);
        }
    }
    // D layout: lane holds rows mtile*16 + 8*hh + j, column nt*16 + r
#pragma unroll
    for (int t = 0; t < 4; ++t) {
        int gcol = (nt0 + t) * 16 + r;
        float* zr = Z + (size_t)(mtile * 16 + 8 * hh) * GCOLS + gcol;
#pragma unroll
        for (int j = 0; j < 8; ++j) zr[(size_t)j * GCOLS] = acc[t][j];
    }
}

// ============================================================================
// Per-(node,head) attention pre-dots: ssrc = z . a[:128], sdst = z . a[128:]
// ============================================================================
__global__ void gat_dots(const float* __restrict__ z, const float* __restrict__ att,
                         float* __restrict__ ssrc, float* __restrict__ sdst,
                         int n_nodes) {
    int gw = (blockIdx.x * blockDim.x + threadIdx.x) >> 5;
    int lane = threadIdx.x & 31;
    if (gw >= n_nodes * GHEADS) return;
    int n = gw >> 2, h = gw & 3;
    const float* zr = z + (size_t)n * GCOLS + h * GHID;
    const float* as = att + h * (2 * GHID);
    const float* ad = as + GHID;
    float s0 = 0.f, s1 = 0.f;
#pragma unroll
    for (int i = lane; i < GHID; i += 32) {
        float v = zr[i];
        s0 += v * as[i];
        s1 += v * ad[i];
    }
#pragma unroll
    for (int m = 16; m; m >>= 1) {
        s0 += __shfl_xor(s0, m, 32);
        s1 += __shfl_xor(s1, m, 32);
    }
    if (lane == 0) { ssrc[gw] = s0; sdst[gw] = s1; }
}

// ============================================================================
// Edge logit + segment max over dst (encoded-uint atomicMax)
// ============================================================================
__global__ void gat_edge_max(const int* __restrict__ src, const int* __restrict__ dst,
                             const float* __restrict__ ssrc, const float* __restrict__ sdst,
                             unsigned* __restrict__ emax, int n_edges) {
    int t = blockIdx.x * blockDim.x + threadIdx.x;
    if (t >= n_edges * GHEADS) return;
    int e = t >> 2, h = t & 3;
    float v = ssrc[src[e] * GHEADS + h] + sdst[dst[e] * GHEADS + h];
    v = (v >= 0.f) ? v : 0.01f * v;                 // leaky_relu
    atomicMax(emax + dst[e] * GHEADS + h, fenc(v));
}

// ============================================================================
// ex = exp(e - emax[dst]); denom[dst] += ex
// ============================================================================
__global__ void gat_edge_exp(const int* __restrict__ src, const int* __restrict__ dst,
                             const float* __restrict__ ssrc, const float* __restrict__ sdst,
                             const unsigned* __restrict__ emax,
                             float* __restrict__ exbuf, float* __restrict__ denom,
                             int n_edges) {
    int t = blockIdx.x * blockDim.x + threadIdx.x;
    if (t >= n_edges * GHEADS) return;
    int e = t >> 2, h = t & 3;
    int d = dst[e];
    float v = ssrc[src[e] * GHEADS + h] + sdst[d * GHEADS + h];
    v = (v >= 0.f) ? v : 0.01f * v;
    float m = fdec(emax[d * GHEADS + h]);
    float ex = __expf(v - m);
    exbuf[t] = ex;
    atomicAdd(denom + d * GHEADS + h, ex);
}

// ============================================================================
// hout[dst] += alpha * z[src]   (one wave per (edge,head), float4 lanes)
// ============================================================================
__global__ void gat_aggregate(const int* __restrict__ src, const int* __restrict__ dst,
                              const float* __restrict__ exbuf, const float* __restrict__ denom,
                              const float* __restrict__ z, float* __restrict__ hout,
                              int n_edges) {
    int gw = (blockIdx.x * blockDim.x + threadIdx.x) >> 5;
    int lane = threadIdx.x & 31;
    if (gw >= n_edges * GHEADS) return;
    int e = gw >> 2, h = gw & 3;
    int s = src[e], d = dst[e];
    float alpha = exbuf[gw] / denom[d * GHEADS + h];
    const float4* zr = (const float4*)(z + (size_t)s * GCOLS + h * GHID);
    float* hr = hout + (size_t)d * GCOLS + h * GHID;
    float4 v = zr[lane];
    int c = lane * 4;
    atomicAdd(hr + c + 0, alpha * v.x);
    atomicAdd(hr + c + 1, alpha * v.y);
    atomicAdd(hr + c + 2, alpha * v.z);
    atomicAdd(hr + c + 3, alpha * v.w);
}

// ============================================================================
// BatchNorm batch stats over N for 512 columns: sum and sumsq of relu(hout)
// ============================================================================
#define BN_ROWS 40
__global__ void gat_bn_stats(const float* __restrict__ hout,
                             float* __restrict__ colsum, float* __restrict__ colsq,
                             int n_nodes) {
    int col = threadIdx.x;                 // blockDim.x == 512
    int row0 = blockIdx.x * BN_ROWS;
    float s = 0.f, q = 0.f;
#pragma unroll
    for (int i = 0; i < BN_ROWS; ++i) {
        int row = row0 + i;
        if (row < n_nodes) {
            float v = hout[(size_t)row * GCOLS + col];
            v = v > 0.f ? v : 0.f;
            s += v; q += v * v;
        }
    }
    atomicAdd(colsum + col, s);
    atomicAdd(colsq + col, q);
}

// ============================================================================
// Per-node: BN normalize, head-score softmax, mix heads -> xout[n,128]
// ============================================================================
__global__ void gat_combine(const float* __restrict__ hout,
                            const float* __restrict__ colsum, const float* __restrict__ colsq,
                            const float* __restrict__ gamma, const float* __restrict__ beta,
                            const float* __restrict__ outw, const float* __restrict__ outb,
                            float* __restrict__ xout, int n_nodes) {
    __shared__ float red[GHEADS][GHID];
    __shared__ float score[GHEADS];
    int n = blockIdx.x;
    int c = threadIdx.x;
    float invN = 1.0f / (float)n_nodes;
    float w = outw[c];
    float y[GHEADS];
#pragma unroll
    for (int h = 0; h < GHEADS; ++h) {
        float v = hout[(size_t)n * GCOLS + h * GHID + c];
        v = v > 0.f ? v : 0.f;
        float mean = colsum[h * GHID + c] * invN;
        float var  = colsq[h * GHID + c] * invN - mean * mean;
        float yy = gamma[h * GHID + c] * (v - mean) * rsqrtf(var + GEPS)
                 + beta[h * GHID + c];
        y[h] = yy;
        red[h][c] = yy * w;
    }
    __syncthreads();
    for (int st = 64; st; st >>= 1) {
        if (c < st) {
#pragma unroll
            for (int h = 0; h < GHEADS; ++h) red[h][c] += red[h][c + st];
        }
        __syncthreads();
    }
    if (c == 0) {
        float b = outb[0];
        float d[GHEADS], mx = -3.4e38f;
#pragma unroll
        for (int h = 0; h < GHEADS; ++h) { d[h] = red[h][0] + b; mx = fmaxf(mx, d[h]); }
        float sum = 0.f;
#pragma unroll
        for (int h = 0; h < GHEADS; ++h) { d[h] = __expf(d[h] - mx); sum += d[h]; }
#pragma unroll
        for (int h = 0; h < GHEADS; ++h) score[h] = d[h] / sum;
    }
    __syncthreads();
    float xv = 0.f;
#pragma unroll
    for (int h = 0; h < GHEADS; ++h) xv += score[h] * y[h];
    xout[(size_t)n * GHID + c] = xv;
}

// ============================================================================
// Graph pooling: per-graph sums + counts, then mean
// ============================================================================
__global__ void graph_acc(const float* __restrict__ x, const int* __restrict__ gid,
                          float* __restrict__ gsum, float* __restrict__ gcnt,
                          int n_nodes) {
    int t = blockIdx.x * blockDim.x + threadIdx.x;
    if (t >= n_nodes * GHID) return;
    int n = t >> 7, c = t & 127;
    int g = gid[n];
    atomicAdd(gsum + g * GHID + c, x[t]);
    if (c == 0) atomicAdd(gcnt + g, 1.0f);
}

__global__ void graph_final(const float* __restrict__ gsum, const float* __restrict__ gcnt,
                            float* __restrict__ out) {
    int t = blockIdx.x * blockDim.x + threadIdx.x;  // G*HID = 16384
    int g = t >> 7;
    out[t] = gsum[t] / fmaxf(gcnt[g], 1.0f);
}

// ============================================================================
// Host-side orchestration
// ============================================================================
extern "C" void kernel_launch(void* const* d_in, const int* in_sizes, int n_in,
                              void* d_out, int out_size, void* d_ws, size_t ws_size,
                              hipStream_t stream) {
    const float* h_in   = (const float*)d_in[0];   // [N, 74]
    const int*   src    = (const int*)  d_in[1];   // [E]
    const int*   dst    = (const int*)  d_in[2];   // [E]
    const int*   gid    = (const int*)  d_in[3];   // [N]
    const float* W0     = (const float*)d_in[4];   // [4, 74, 128]
    const float* Wr     = (const float*)d_in[5];   // [1, 4, 128, 128]
    const float* att    = (const float*)d_in[6];   // [2, 4, 256]
    const float* gamma  = (const float*)d_in[7];   // [2, 4, 128]
    const float* beta   = (const float*)d_in[8];   // [2, 4, 128]
    const float* outw   = (const float*)d_in[9];   // [2, 128]
    const float* outb   = (const float*)d_in[10];  // [2]
    float* out = (float*)d_out;                    // [G, 128]

    // ---- workspace layout (floats, then f16 regions; all 32B aligned) ----
    float* ws = (float*)d_ws;
    float*    z      = ws;                                      // N*512
    float*    hout   = z      + (size_t)GN * GCOLS;             // N*512
    float*    x1     = hout   + (size_t)GN * GCOLS;             // N*128
    float*    x2     = x1     + (size_t)GN * GHID;              // N*128
    float*    ssrc   = x2     + (size_t)GN * GHID;              // N*4
    float*    sdst   = ssrc   + (size_t)GN * GHEADS;            // N*4
    unsigned* emax   = (unsigned*)(sdst + (size_t)GN * GHEADS); // N*4
    float*    denom  = (float*)emax + (size_t)GN * GHEADS;      // N*4
    float*    exbuf  = denom  + (size_t)GN * GHEADS;            // E*4
    float*    colsum = exbuf  + (size_t)GE * GHEADS;            // 512
    float*    colsq  = colsum + GCOLS;                          // 512
    float*    gsum   = colsq  + GCOLS;                          // G*128
    float*    gcnt   = gsum   + (size_t)GG * GHID;              // G
    _Float16* Xh     = (_Float16*)(gcnt + GG);                  // N*128 halves max
    _Float16* Wp     = Xh + (size_t)GN * GHID;                  // 512*128 halves max

    const int nbDots  = (GN * GHEADS * 32 + 255) / 256;   // wave per (node,head)
    const int nbEdge  = (GE * GHEADS + 255) / 256;        // thread per (edge,head)
    const int nbAgg   = (GE * GHEADS * 32 + 255) / 256;   // wave per (edge,head)
    const int nbStats = (GN + BN_ROWS - 1) / BN_ROWS;
    const int nbGAcc  = (GN * GHID + 255) / 256;
    const int Mtiles  = GN / 16;                          // 1250 exactly
    const dim3 gemmGrid((Mtiles + 7) / 8, GCOLS / 64);    // 8 waves/block, 4 ntiles/wave

    const float* xin = h_in;
    for (int layer = 0; layer < 2; ++layer) {
        const int K    = (layer == 0) ? GIN : GHID;
        const int Kpad = (K + 31) & ~31;                  // 96 / 128
        const float* Wl = (layer == 0) ? W0 : Wr;
        float* xout = (layer == 0) ? x1 : x2;

        // zero per-layer accumulators
        hipMemsetAsync(hout,   0, (size_t)GN * GCOLS * sizeof(float), stream);
        hipMemsetAsync(emax,   0, (size_t)GN * GHEADS * sizeof(unsigned), stream);
        hipMemsetAsync(denom,  0, (size_t)GN * GHEADS * sizeof(float), stream);
        hipMemsetAsync(colsum, 0, 2 * GCOLS * sizeof(float), stream);  // colsum+colsq

        // pack operands to f16 (bounds logic only here)
        pack_x_f16<<<GN, 128, 0, stream>>>(xin, Xh, K, Kpad);
        const int wtot = 32 * (Kpad >> 5) * 512;
        pack_w_f16<<<(wtot + 255) / 256, 256, 0, stream>>>(Wl, Wp, K, Kpad);

        gat_gemm_wmma<<<gemmGrid, 256, 0, stream>>>(Xh, Wp, z, Mtiles, Kpad);
        gat_dots<<<nbDots, 256, 0, stream>>>(z, att + layer * GHEADS * 2 * GHID,
                                             ssrc, sdst, GN);
        gat_edge_max<<<nbEdge, 256, 0, stream>>>(src, dst, ssrc, sdst, emax, GE);
        gat_edge_exp<<<nbEdge, 256, 0, stream>>>(src, dst, ssrc, sdst, emax,
                                                 exbuf, denom, GE);
        gat_aggregate<<<nbAgg, 256, 0, stream>>>(src, dst, exbuf, denom, z, hout, GE);
        gat_bn_stats<<<nbStats, 512, 0, stream>>>(hout, colsum, colsq, GN);
        gat_combine<<<GN, GHID, 0, stream>>>(hout, colsum, colsq,
                                             gamma + layer * GHEADS * GHID,
                                             beta  + layer * GHEADS * GHID,
                                             outw  + layer * GHID,
                                             outb  + layer,
                                             xout, GN);
        xin = xout;
    }

    // graph mean pooling
    hipMemsetAsync(gsum, 0, ((size_t)GG * GHID + GG) * sizeof(float), stream);
    graph_acc<<<nbGAcc, 256, 0, stream>>>(x2, gid, gsum, gcnt, GN);
    graph_final<<<(GG * GHID + 255) / 256, 256, 0, stream>>>(gsum, gcnt, out);
}
